// Grok1MoEUnfused_25469156065341
// MI455X (gfx1250) — compile-verified, matrix operand order checked by reading, not run on hardware
//
#include <hip/hip_runtime.h>
#include <hip/hip_bf16.h>
#include <math.h>

#define T_TOK 2048
#define HDIM  1024
#define FDIM  4096
#define NEXP  8
#define PAIRS (T_TOK * 2)

typedef __attribute__((ext_vector_type(16))) __bf16 v16bf;
typedef __attribute__((ext_vector_type(8)))  float  v8f;
typedef __attribute__((ext_vector_type(4)))  float  v4f;
typedef __attribute__((ext_vector_type(4)))  __bf16 v4bf;
typedef __attribute__((ext_vector_type(4)))  int    v4i;

union BFrag { v16bf v; uint4 q[2]; };

__device__ __forceinline__ void cvt_store4(__bf16* d, float4 x) {
    v4f f; f[0] = x.x; f[1] = x.y; f[2] = x.z; f[3] = x.w;
    *(v4bf*)d = __builtin_convertvector(f, v4bf);   // -> v_cvt_pk_bf16_f32 pair
}

// -------- optional CDNA5 async global->LDS copy path (ASYNCcnt tracked) -----
#if __has_builtin(__builtin_amdgcn_global_load_async_to_lds_b128)
#define ASYNC_CP 1
typedef __attribute__((address_space(1))) v4i as1_v4i;
typedef __attribute__((address_space(3))) v4i as3_v4i;
__device__ __forceinline__ void async_cp16(const void* g, void* l) {
    __builtin_amdgcn_global_load_async_to_lds_b128(
        (as1_v4i*)(size_t)g, (as3_v4i*)(size_t)l, 0, 0);
}
__device__ __forceinline__ void async_wait0() {
#if __has_builtin(__builtin_amdgcn_s_wait_asynccnt)
    __builtin_amdgcn_s_wait_asynccnt(0);
#else
    asm volatile("s_wait_asynccnt 0x0" ::: "memory");
#endif
}
#else
#define ASYNC_CP 0
__device__ __forceinline__ void async_cp16(const void* g, void* l) {
    *(uint4*)l = *(const uint4*)g;
}
__device__ __forceinline__ void async_wait0() {}
#endif

// ---------------------------------------------------------------- router ----
__global__ __launch_bounds__(256) void router_kernel(
    const float* __restrict__ hs, const float* __restrict__ gw,
    int* __restrict__ top2_id, float* __restrict__ top2_w) {
    __shared__ float sgw[NEXP * HDIM];
    const int tid = threadIdx.x;
    const float4* g4 = (const float4*)gw;
    float4* s4 = (float4*)sgw;
    for (int i = tid; i < NEXP * HDIM / 4; i += 256) s4[i] = g4[i];
    __syncthreads();

    const int t = blockIdx.x * 256 + tid;
    float acc[NEXP];
#pragma unroll
    for (int e = 0; e < NEXP; ++e) acc[e] = 0.f;
    const float4* h4 = (const float4*)(hs + (size_t)t * HDIM);
    for (int q = 0; q < HDIM / 4; ++q) {
        float4 x = h4[q];
        const int hb = q * 4;
#pragma unroll
        for (int e = 0; e < NEXP; ++e) {
            const float* ge = sgw + e * HDIM + hb;
            acc[e] += x.x * ge[0] + x.y * ge[1] + x.z * ge[2] + x.w * ge[3];
        }
    }
    float mx = -1e30f;
#pragma unroll
    for (int e = 0; e < NEXP; ++e) {
        acc[e] = 30.f * tanhf(acc[e] * (1.f / 30.f));
        mx = fmaxf(mx, acc[e]);
    }
    float sum = 0.f;
#pragma unroll
    for (int e = 0; e < NEXP; ++e) { acc[e] = expf(acc[e] - mx); sum += acc[e]; }
    const float inv = 1.f / sum;
#pragma unroll
    for (int e = 0; e < NEXP; ++e) acc[e] *= inv;

    int i0 = 0; float v0 = acc[0];
#pragma unroll
    for (int e = 1; e < NEXP; ++e) if (acc[e] > v0) { v0 = acc[e]; i0 = e; }
    int i1 = -1; float v1 = -1.f;
#pragma unroll
    for (int e = 0; e < NEXP; ++e) if (e != i0 && acc[e] > v1) { v1 = acc[e]; i1 = e; }
    top2_id[t * 2 + 0] = i0; top2_w[t * 2 + 0] = v0;
    top2_id[t * 2 + 1] = i1; top2_w[t * 2 + 1] = v1;
}

// ------------------------------------------------------ stable compaction ---
__global__ __launch_bounds__(256) void build_lists_kernel(
    const int* __restrict__ top2_id, int* __restrict__ counts, int* __restrict__ offs,
    int* __restrict__ list_pair, int* __restrict__ slot_of_pair) {
    __shared__ int sCnt[NEXP][256];
    __shared__ int sOff[NEXP];
    const int tid = threadIdx.x;
    const int CH = PAIRS / 256;
    int lc[NEXP];
#pragma unroll
    for (int e = 0; e < NEXP; ++e) lc[e] = 0;
    const int base = tid * CH;
    for (int i = 0; i < CH; ++i) lc[top2_id[base + i]]++;
#pragma unroll
    for (int e = 0; e < NEXP; ++e) sCnt[e][tid] = lc[e];
    __syncthreads();
    if (tid < NEXP) {
        int run = 0;
        for (int i = 0; i < 256; ++i) { int c = sCnt[tid][i]; sCnt[tid][i] = run; run += c; }
        counts[tid] = run;
    }
    __syncthreads();
    if (tid == 0) {
        int run = 0;
        for (int e = 0; e < NEXP; ++e) { offs[e] = run; sOff[e] = run; run += counts[e]; }
        offs[NEXP] = run;
    }
    __syncthreads();
    int pos[NEXP];
#pragma unroll
    for (int e = 0; e < NEXP; ++e) pos[e] = sOff[e] + sCnt[e][tid];
    for (int i = 0; i < CH; ++i) {
        const int p = base + i;
        const int e = top2_id[p];
        const int s = pos[e]++;
        list_pair[s] = p;
        slot_of_pair[p] = s;
    }
}

// ------------------------------------------------ hidden fp32 -> bf16 once --
__global__ __launch_bounds__(256) void cvt_hidden_kernel(
    const float* __restrict__ hs, __bf16* __restrict__ hsb) {
    const int i = (blockIdx.x * 256 + threadIdx.x) * 4;
    float4 x = *(const float4*)(hs + i);
    v4f f; f[0] = x.x; f[1] = x.y; f[2] = x.z; f[3] = x.w;
    *(v4bf*)(hsb + i) = __builtin_convertvector(f, v4bf);
}

// ------------------------------------------- GEMM1: gelu(x@w1^T) * (x@w3^T) -
// grid: (F/64, 16, E); tile M=128, N=64, K=32
__global__ __launch_bounds__(256) void gemm1_kernel(
    const __bf16* __restrict__ hsb, const float* __restrict__ w1, const float* __restrict__ w3,
    const int* __restrict__ list_pair, const int* __restrict__ offs,
    __bf16* __restrict__ act) {
    __shared__ __align__(16) __bf16 sA[128][48];
    __shared__ __align__(16) __bf16 sB1[64][48];
    __shared__ __align__(16) __bf16 sB3[64][48];
    __shared__ int sTok[128];

    const int e  = blockIdx.z;
    const int m0 = blockIdx.y * 128;
    const int n0 = blockIdx.x * 64;
    const int base = offs[e];
    const int cnt  = offs[e + 1] - base;
    if (m0 >= cnt) return;

    const int tid  = threadIdx.x;
    const int wid  = tid >> 5;
    const int lane = tid & 31;
    const int waveM = wid & 3;    // 4 waves over M: 32 rows each
    const int waveN = wid >> 2;   // 2 waves over N: 32 cols each
    const int lrow = lane & 15;
    const int kh   = (lane >> 4) * 8;

    if (tid < 128) {
        // clamp padded rows to token 0: loads stay in-bounds, results unused
        sTok[tid] = (m0 + tid < cnt) ? (list_pair[base + m0 + tid] >> 1) : 0;
    }

    v8f ag[2][2], au[2][2];
#pragma unroll
    for (int i = 0; i < 2; ++i)
#pragma unroll
        for (int j = 0; j < 2; ++j) { ag[i][j] = (v8f){}; au[i][j] = (v8f){}; }

    __syncthreads();

    for (int k0 = 0; k0 < HDIM; k0 += 32) {
#pragma unroll
        for (int it = 0; it < 2; ++it) {        // A: gathered bf16 rows, 16B copies
            const int idx = tid + it * 256;
            const int row = idx >> 2, c8 = (idx & 3) * 8;
            async_cp16(hsb + (size_t)sTok[row] * HDIM + k0 + c8, &sA[row][c8]);
        }
#pragma unroll
        for (int it = 0; it < 2; ++it) {        // B1/B3: 64x32 fp32 -> bf16
            const int idx = tid + it * 256;
            const int row = idx >> 3, c4 = (idx & 7) * 4;
            const size_t boff = ((size_t)e * FDIM + (n0 + row)) * HDIM + k0 + c4;
            cvt_store4(&sB1[row][c4], *(const float4*)(w1 + boff));
            cvt_store4(&sB3[row][c4], *(const float4*)(w3 + boff));
        }
        async_wait0();
        __syncthreads();

        BFrag a[2], b1[2], b3[2];
#pragma unroll
        for (int i = 0; i < 2; ++i) {
            const __bf16* p = &sA[waveM * 32 + i * 16 + lrow][kh];
            a[i].q[0] = *(const uint4*)p;
            a[i].q[1] = *(const uint4*)(p + 16);
        }
#pragma unroll
        for (int j = 0; j < 2; ++j) {
            const __bf16* p1 = &sB1[waveN * 32 + j * 16 + lrow][kh];
            b1[j].q[0] = *(const uint4*)p1; b1[j].q[1] = *(const uint4*)(p1 + 16);
            const __bf16* p3 = &sB3[waveN * 32 + j * 16 + lrow][kh];
            b3[j].q[0] = *(const uint4*)p3; b3[j].q[1] = *(const uint4*)(p3 + 16);
        }
#pragma unroll
        for (int i = 0; i < 2; ++i)
#pragma unroll
            for (int j = 0; j < 2; ++j) {
                ag[i][j] = __builtin_amdgcn_wmma_f32_16x16x32_bf16(
                    false, a[i].v, false, b1[j].v, (short)0, ag[i][j], false, false);
                au[i][j] = __builtin_amdgcn_wmma_f32_16x16x32_bf16(
                    false, a[i].v, false, b3[j].v, (short)0, au[i][j], false, false);
            }
        __syncthreads();
    }

    const int mhalf = (lane >> 4) * 8;
    const int ncol  = lane & 15;
#pragma unroll
    for (int i = 0; i < 2; ++i)
#pragma unroll
        for (int j = 0; j < 2; ++j)
#pragma unroll
            for (int v = 0; v < 8; ++v) {
                const int m = waveM * 32 + i * 16 + v + mhalf;
                if (m0 + m < cnt) {
                    const float g = ag[i][j][v];
                    const float u = au[i][j][v];
                    const float gel = 0.5f * g * (1.0f + erff(g * 0.70710678118f));
                    act[(size_t)(base + m0 + m) * FDIM + (n0 + waveN * 32 + j * 16 + ncol)]
                        = (__bf16)(gel * u);
                }
            }
}

// ------------------------------------------------- GEMM2: (g*u) @ w2^T ------
// grid: (H/128, 16, E); tile M=128, N=128, K=32
__global__ __launch_bounds__(256) void gemm2_kernel(
    const __bf16* __restrict__ act, const float* __restrict__ w2,
    const int* __restrict__ offs, float* __restrict__ pairout) {
    __shared__ __align__(16) __bf16 sA[128][48];
    __shared__ __align__(16) __bf16 sB[128][48];

    const int e  = blockIdx.z;
    const int m0 = blockIdx.y * 128;
    const int n0 = blockIdx.x * 128;
    const int base = offs[e];
    const int cnt  = offs[e + 1] - base;
    if (m0 >= cnt) return;

    const int tid  = threadIdx.x;
    const int wid  = tid >> 5;
    const int lane = tid & 31;
    const int waveM = wid & 1;    // 2 waves over M: 64 rows (4 frags)
    const int waveN = wid >> 1;   // 4 waves over N: 32 cols (2 frags)
    const int lrow = lane & 15;
    const int kh   = (lane >> 4) * 8;

    v8f acc[4][2];
#pragma unroll
    for (int i = 0; i < 4; ++i)
#pragma unroll
        for (int j = 0; j < 2; ++j) acc[i][j] = (v8f){};

    for (int k0 = 0; k0 < FDIM; k0 += 32) {
#pragma unroll
        for (int it = 0; it < 2; ++it) {        // A: bf16 contiguous, 16B copies
            const int idx = tid + it * 256;
            const int row = idx >> 2, c8 = (idx & 3) * 8;
            async_cp16(act + (size_t)(base + m0 + row) * FDIM + k0 + c8, &sA[row][c8]);
        }
#pragma unroll
        for (int it = 0; it < 4; ++it) {        // B: w2 fp32 128x32 -> bf16
            const int idx = tid + it * 256;
            const int row = idx >> 3, c4 = (idx & 7) * 4;
            cvt_store4(&sB[row][c4],
                       *(const float4*)(w2 + ((size_t)e * HDIM + (n0 + row)) * FDIM + k0 + c4));
        }
        async_wait0();
        __syncthreads();

        BFrag a[4], b[2];
#pragma unroll
        for (int i = 0; i < 4; ++i) {
            const __bf16* p = &sA[waveM * 64 + i * 16 + lrow][kh];
            a[i].q[0] = *(const uint4*)p;
            a[i].q[1] = *(const uint4*)(p + 16);
        }
#pragma unroll
        for (int j = 0; j < 2; ++j) {
            const __bf16* p = &sB[waveN * 32 + j * 16 + lrow][kh];
            b[j].q[0] = *(const uint4*)p;
            b[j].q[1] = *(const uint4*)(p + 16);
        }
#pragma unroll
        for (int i = 0; i < 4; ++i)
#pragma unroll
            for (int j = 0; j < 2; ++j)
                acc[i][j] = __builtin_amdgcn_wmma_f32_16x16x32_bf16(
                    false, a[i].v, false, b[j].v, (short)0, acc[i][j], false, false);
        __syncthreads();
    }

    const int mhalf = (lane >> 4) * 8;
    const int ncol  = lane & 15;
#pragma unroll
    for (int i = 0; i < 4; ++i)
#pragma unroll
        for (int j = 0; j < 2; ++j)
#pragma unroll
            for (int v = 0; v < 8; ++v) {
                const int m = waveM * 64 + i * 16 + v + mhalf;
                if (m0 + m < cnt)
                    pairout[(size_t)(base + m0 + m) * HDIM + (n0 + waveN * 32 + j * 16 + ncol)]
                        = acc[i][j][v];
            }
}

// ---------------------------------------------------------------- combine ---
__global__ __launch_bounds__(256) void combine_kernel(
    const float* __restrict__ pairout, const int* __restrict__ slot_of_pair,
    const float* __restrict__ top2_w, float* __restrict__ out) {
    const int idx = blockIdx.x * 256 + threadIdx.x;       // over T*H/4
    const int t  = idx / (HDIM / 4);
    const int hc = (idx % (HDIM / 4)) * 4;
    const int s0 = slot_of_pair[2 * t], s1 = slot_of_pair[2 * t + 1];
    const float w0 = top2_w[2 * t], w1v = top2_w[2 * t + 1];
    float4 a = *(const float4*)(pairout + (size_t)s0 * HDIM + hc);
    float4 b = *(const float4*)(pairout + (size_t)s1 * HDIM + hc);
    float4 o;
    o.x = w0 * a.x + w1v * b.x;
    o.y = w0 * a.y + w1v * b.y;
    o.z = w0 * a.z + w1v * b.z;
    o.w = w0 * a.w + w1v * b.w;
    *(float4*)(out + (size_t)t * HDIM + hc) = o;
}

// ---------------------------------------------------------------- launch ----
extern "C" void kernel_launch(void* const* d_in, const int* in_sizes, int n_in,
                              void* d_out, int out_size, void* d_ws, size_t ws_size,
                              hipStream_t stream) {
    const float* hs = (const float*)d_in[0];
    const float* gw = (const float*)d_in[1];
    const float* w1 = (const float*)d_in[2];
    const float* w2 = (const float*)d_in[3];
    const float* w3 = (const float*)d_in[4];

    char* ws = (char*)d_ws;
    size_t off = 0;
    auto alloc = [&](size_t bytes) -> char* {
        char* p = ws + off;
        off += (bytes + 255) & ~(size_t)255;
        return p;
    };
    int*    top2_id      = (int*)alloc((size_t)PAIRS * 4);
    float*  top2_w       = (float*)alloc((size_t)PAIRS * 4);
    int*    counts       = (int*)alloc(NEXP * 4);
    int*    offs         = (int*)alloc((NEXP + 1) * 4);
    int*    list_pair    = (int*)alloc((size_t)PAIRS * 4);
    int*    slot_of_pair = (int*)alloc((size_t)PAIRS * 4);
    __bf16* hsb          = (__bf16*)alloc((size_t)T_TOK * HDIM * 2);
    __bf16* act          = (__bf16*)alloc((size_t)(PAIRS + 128) * FDIM * 2);
    float*  pairout      = (float*)alloc((size_t)PAIRS * HDIM * 4);

    router_kernel<<<T_TOK / 256, 256, 0, stream>>>(hs, gw, top2_id, top2_w);
    build_lists_kernel<<<1, 256, 0, stream>>>(top2_id, counts, offs, list_pair, slot_of_pair);
    cvt_hidden_kernel<<<(T_TOK * HDIM / 4) / 256, 256, 0, stream>>>(hs, hsb);
    gemm1_kernel<<<dim3(FDIM / 64, T_TOK / 128, NEXP), 256, 0, stream>>>(
        hsb, w1, w3, list_pair, offs, act);
    gemm2_kernel<<<dim3(HDIM / 128, T_TOK / 128, NEXP), 256, 0, stream>>>(
        act, w2, offs, pairout);
    combine_kernel<<<(T_TOK * HDIM / 4) / 256, 256, 0, stream>>>(
        pairout, slot_of_pair, top2_w, (float*)d_out);
}
